// GatingNetwork_20873541059273
// MI455X (gfx1250) — matrix operand order repaired and checked by reading, not compiled, not run on hardware
//
#include <hip/hip_runtime.h>
#include <hip/hip_bf16.h>

// Problem constants (match reference)
#define BSZ 16384
#define DDIM 4096
#define HDIM 2048
#define EEXP 2
#define TEMP 0.8f

typedef __attribute__((ext_vector_type(16))) __bf16 v16bf;
typedef __attribute__((ext_vector_type(8)))  float  v8f;

union BF16Frag {
    uint4 q[2];
    v16bf v;
};

__device__ __forceinline__ unsigned short f2bf(float f) {
    unsigned u = __float_as_uint(f);
    unsigned r = u + 0x7FFFu + ((u >> 16) & 1u);   // round-to-nearest-even
    return (unsigned short)(r >> 16);
}
__device__ __forceinline__ unsigned pk2(float a, float b) {
    return (unsigned)f2bf(a) | ((unsigned)f2bf(b) << 16);
}

// ---------------------------------------------------------------------------
// Kernel 0: zero logits accumulator + entropy accumulator in workspace
// ---------------------------------------------------------------------------
__global__ void zero_ws_kernel(float* p, int n) {
    int i = blockIdx.x * blockDim.x + threadIdx.x;
    if (i < n) p[i] = 0.0f;
}

// ---------------------------------------------------------------------------
// Kernel 0b: one-time W1 fp32 -> bf16 (row-major [H][D]) into workspace.
// Removes per-B-tile redundant conversion; bf16 W1 (16MB) then lives in L2.
// ---------------------------------------------------------------------------
__global__ void __launch_bounds__(256)
cvt_w1_kernel(const float* __restrict__ W1, unsigned* __restrict__ W1b) {
    const size_t i = (size_t)(blockIdx.x * 256 + threadIdx.x) * 8;  // 8 floats
    if (i >= (size_t)HDIM * DDIM) return;
    const float4* g = (const float4*)(W1 + i);
    float4 a = g[0], b = g[1];
    uint4 o;
    o.x = pk2(a.x, a.y); o.y = pk2(a.z, a.w);
    o.z = pk2(b.x, b.y); o.w = pk2(b.z, b.w);
    *(uint4*)(W1b + i / 2) = o;
}

// ---------------------------------------------------------------------------
// Kernel 1: fused GEMM1 (bf16 WMMA) + bias + ReLU + partial router logits.
//   Block: 128x128 tile of h = x*W1^T. 8 waves in 4x2 grid, each wave owns
//   32 rows x 64 cols = 2x4 WMMA tiles. K staged 64 at a time:
//     - x tile: fp32 global load -> VALU bf16 convert -> ds_store
//     - W1 tile: bf16 already in memory -> global_load_async_to_lds_b128
//   so the weight DMA overlaps the activation conversion.
// ---------------------------------------------------------------------------
#define ROWP 72   // LDS row pitch in ushorts (64 data + 8 pad = 144 bytes)

#define ASYNC_CP(OFF)                                                        \
    asm volatile("global_load_async_to_lds_b128 %0, %1, %2 offset:" #OFF     \
                 :: "v"(ldsoff), "v"(voff), "s"(W1b) : "memory")

__global__ void __launch_bounds__(256)
router_gemm_kernel(const float* __restrict__ x,
                   const unsigned short* __restrict__ W1b,   // bf16 [H][D]
                   const float* __restrict__ b1,
                   const float* __restrict__ W2,
                   float* __restrict__ logits) {
    __shared__ unsigned short ldsA[128 * ROWP];
    __shared__ unsigned short ldsB[128 * ROWP];

    const int h0   = blockIdx.x * 128;   // hidden-tile origin (fast-varying:
    const int row0 = blockIdx.y * 128;   // 16 consecutive blocks share x rows)

    const int t    = threadIdx.x;
    const int lane = t & 31;
    const int wid  = t >> 5;
    const int lh   = lane & 15;          // lane within half
    const int half = lane >> 4;          // 0: lanes 0-15, 1: lanes 16-31
    const int m0w  = (wid & 3) * 32;     // wave row offset inside tile
    const int c0w  = (wid >> 2) * 64;    // wave col offset inside tile

    // loader assignment: 2 threads per row, each covers 32 K elements
    const int lrow = t >> 1;
    const int kp   = (t & 1) * 32;

    v8f acc[2][4];
    #pragma unroll
    for (int rt = 0; rt < 2; ++rt)
        #pragma unroll
        for (int ct = 0; ct < 4; ++ct)
            #pragma unroll
            for (int i = 0; i < 8; ++i)
                acc[rt][ct][i] = 0.0f;

    const unsigned ldsoff = (unsigned)(uintptr_t)&ldsB[lrow * ROWP + kp];

    const int KSTEPS = DDIM / 64;
    for (int kt = 0; kt < KSTEPS; ++kt) {
        const int k0 = kt * 64;
        __syncthreads();   // previous iteration finished reading LDS

        // ---- W1 bf16 tile: async DMA global -> LDS (64B per thread) ----
        {
            const unsigned voff =
                (unsigned)(((unsigned)(h0 + lrow) * DDIM + k0 + kp) * 2);
            ASYNC_CP(0);
            ASYNC_CP(16);
            ASYNC_CP(32);
            ASYNC_CP(48);
        }

        // ---- x tile: fp32 load -> bf16 convert -> LDS ------------------
        {
            const float4* ga =
                (const float4*)(x + (size_t)(row0 + lrow) * DDIM + k0 + kp);
            float4 f[8];
            #pragma unroll
            for (int i = 0; i < 8; ++i) f[i] = ga[i];
            unsigned w[16];
            #pragma unroll
            for (int i = 0; i < 8; ++i) {
                w[2 * i + 0] = pk2(f[i].x, f[i].y);
                w[2 * i + 1] = pk2(f[i].z, f[i].w);
            }
            uint4* da = (uint4*)&ldsA[lrow * ROWP + kp];
            da[0] = *(const uint4*)&w[0];
            da[1] = *(const uint4*)&w[4];
            da[2] = *(const uint4*)&w[8];
            da[3] = *(const uint4*)&w[12];
        }

        asm volatile("s_wait_asynccnt 0x0" ::: "memory");
        __syncthreads();   // tiles visible

        // ---- 2 K-chunks of 32: fragment loads + 8 WMMAs each -----------
        #pragma unroll
        for (int kc = 0; kc < 2; ++kc) {
            BF16Frag afr[2];
            #pragma unroll
            for (int rt = 0; rt < 2; ++rt) {
                // A 16x32 bf16: lane holds K in {h*8..h*8+7} and {16+h*8..}
                const unsigned short* p =
                    &ldsA[(m0w + rt * 16 + lh) * ROWP + kc * 32 + half * 8];
                afr[rt].q[0] = *(const uint4*)p;
                afr[rt].q[1] = *(const uint4*)(p + 16);
            }
            BF16Frag bfr[4];
            #pragma unroll
            for (int ct = 0; ct < 4; ++ct) {
                // B 32x16 bf16: lane holds K in {h*8..h*8+7} and {16+h*8..}
                const unsigned short* p =
                    &ldsB[(c0w + ct * 16 + lh) * ROWP + kc * 32 + half * 8];
                bfr[ct].q[0] = *(const uint4*)p;
                bfr[ct].q[1] = *(const uint4*)(p + 16);
            }
            #pragma unroll
            for (int rt = 0; rt < 2; ++rt)
                #pragma unroll
                for (int ct = 0; ct < 4; ++ct)
                    acc[rt][ct] = __builtin_amdgcn_wmma_f32_16x16x32_bf16(
                        false, afr[rt].v, false, bfr[ct].v,
                        (short)0, acc[rt][ct], false, false);
        }
    }

    // ---- fused epilogue: relu(h + b1) * W2^T -> partial logits ---------
    float w2a[4], w2b[4], b1v[4];
    #pragma unroll
    for (int ct = 0; ct < 4; ++ct) {
        const int col = h0 + c0w + ct * 16 + lh;
        b1v[ct] = b1[col];
        w2a[ct] = W2[col];             // W2[0, col]
        w2b[ct] = W2[HDIM + col];      // W2[1, col]
    }

    #pragma unroll
    for (int rt = 0; rt < 2; ++rt) {
        float pl0[8], pl1[8];
        #pragma unroll
        for (int v = 0; v < 8; ++v) { pl0[v] = 0.0f; pl1[v] = 0.0f; }
        #pragma unroll
        for (int ct = 0; ct < 4; ++ct) {
            #pragma unroll
            for (int v = 0; v < 8; ++v) {
                float r = fmaxf(acc[rt][ct][v] + b1v[ct], 0.0f);
                pl0[v] += r * w2a[ct];
                pl1[v] += r * w2b[ct];
            }
        }
        // reduce over the 16 lanes of each half (halves hold disjoint rows)
        #pragma unroll
        for (int m = 1; m < 16; m <<= 1) {
            #pragma unroll
            for (int v = 0; v < 8; ++v) {
                pl0[v] += __shfl_xor(pl0[v], m, 32);
                pl1[v] += __shfl_xor(pl1[v], m, 32);
            }
        }
        if (lh == 0) {
            const int rowbase = row0 + m0w + rt * 16 + half * 8;
            #pragma unroll
            for (int v = 0; v < 8; ++v) {
                atomicAdd(&logits[(size_t)(rowbase + v) * 2 + 0], pl0[v]);
                atomicAdd(&logits[(size_t)(rowbase + v) * 2 + 1], pl1[v]);
            }
        }
    }
}

// ---------------------------------------------------------------------------
// Kernel 2: per-row temperature softmax, top-k (E=K=2), entropy accumulation
// ---------------------------------------------------------------------------
__global__ void __launch_bounds__(256)
router_softmax_kernel(const float* __restrict__ logits,
                      const float* __restrict__ b2,
                      float* __restrict__ out_w,
                      float* __restrict__ out_idx,
                      float* __restrict__ ent_acc) {
    const int b = blockIdx.x * blockDim.x + threadIdx.x;

    float ent = 0.0f;
    if (b < BSZ) {
        const float l0 = logits[(size_t)b * 2 + 0] + b2[0];
        const float l1 = logits[(size_t)b * 2 + 1] + b2[1];

        // temperature-scaled softmax (sum is exactly 1 for E=2)
        const float s0 = l0 / TEMP, s1 = l1 / TEMP;
        const float ms = fmaxf(s0, s1);
        const float e0 = __expf(s0 - ms), e1 = __expf(s1 - ms);
        const float inv = 1.0f / (e0 + e1);
        const float p0 = e0 * inv, p1 = e1 * inv;

        float w0, w1, i0, i1;
        if (p0 >= p1) { w0 = p0; w1 = p1; i0 = 0.0f; i1 = 1.0f; }
        else          { w0 = p1; w1 = p0; i0 = 1.0f; i1 = 0.0f; }
        const float wn = 1.0f / (w0 + w1);
        out_w[(size_t)b * 2 + 0]   = w0 * wn;
        out_w[(size_t)b * 2 + 1]   = w1 * wn;
        out_idx[(size_t)b * 2 + 0] = i0;
        out_idx[(size_t)b * 2 + 1] = i1;

        // entropy from UN-scaled logits
        const float mu = fmaxf(l0, l1);
        const float q0e = __expf(l0 - mu), q1e = __expf(l1 - mu);
        const float qinv = 1.0f / (q0e + q1e);
        const float q0 = q0e * qinv, q1 = q1e * qinv;
        ent = -(q0 * __logf(q0 + 1e-10f) + q1 * __logf(q1 + 1e-10f));
    }

    // wave then block reduction, one atomic per block
    #pragma unroll
    for (int m = 1; m < 32; m <<= 1) ent += __shfl_xor(ent, m, 32);
    __shared__ float wsum[8];
    const int lane = threadIdx.x & 31, wid = threadIdx.x >> 5;
    if (lane == 0) wsum[wid] = ent;
    __syncthreads();
    if (threadIdx.x == 0) {
        float s = 0.0f;
        #pragma unroll
        for (int i = 0; i < 8; ++i) s += wsum[i];
        atomicAdd(ent_acc, s);
    }
}

// ---------------------------------------------------------------------------
// Kernel 3: finalize uncertainty scalar
// ---------------------------------------------------------------------------
__global__ void finalize_kernel(const float* __restrict__ ent_acc,
                                float* __restrict__ out_unc) {
    out_unc[0] = (ent_acc[0] / (float)BSZ) / __logf((float)EEXP);
}

// ---------------------------------------------------------------------------
extern "C" void kernel_launch(void* const* d_in, const int* in_sizes, int n_in,
                              void* d_out, int out_size, void* d_ws, size_t ws_size,
                              hipStream_t stream) {
    const float* x  = (const float*)d_in[0];
    const float* W1 = (const float*)d_in[1];
    const float* b1 = (const float*)d_in[2];
    const float* W2 = (const float*)d_in[3];
    const float* b2 = (const float*)d_in[4];

    float* out     = (float*)d_out;
    float* out_w   = out;                 // [B,2] routing weights
    float* out_idx = out + 2 * BSZ;       // [B,2] expert indices (as float)
    float* out_unc = out + 4 * BSZ;       // scalar uncertainty

    // workspace layout: logits [B,2] | ent scalar | pad | W1 bf16 [H][D]
    float* logits  = (float*)d_ws;
    float* ent_acc = logits + 2 * BSZ;
    unsigned short* W1b =
        (unsigned short*)((char*)d_ws + ((2 * BSZ + 1) * 4 + 255) / 256 * 256);
    const int nzero = 2 * BSZ + 1;

    zero_ws_kernel<<<(nzero + 255) / 256, 256, 0, stream>>>(logits, nzero);

    cvt_w1_kernel<<<(HDIM * DDIM / 8 + 255) / 256, 256, 0, stream>>>(
        W1, (unsigned*)W1b);

    dim3 grid(HDIM / 128, BSZ / 128);     // H-tiles fast -> x L2 reuse
    router_gemm_kernel<<<grid, 256, 0, stream>>>(x, W1b, b1, W2, logits);

    router_softmax_kernel<<<(BSZ + 255) / 256, 256, 0, stream>>>(
        logits, b2, out_w, out_idx, ent_acc);

    finalize_kernel<<<1, 1, 0, stream>>>(ent_acc, out_unc);
}
